// TernaryLinear_18786186952995
// MI455X (gfx1250) — matrix operand order repaired and checked by reading, not compiled, not run on hardware
//
#include <hip/hip_runtime.h>

typedef __attribute__((ext_vector_type(16))) _Float16     v16h;
typedef __attribute__((ext_vector_type(8)))  float        v8f;
typedef __attribute__((ext_vector_type(4)))  unsigned int v4u;

#define IN_F   4096
#define OUT_F  11008
#define TOK    32
#define NTILES (OUT_F / 16)          // 688 column tiles
#define KSEG   8                     // K split factor for occupancy
#define KSEG_LEN (IN_F / KSEG)       // 512
#define GEMM_BLOCK 128               // 4 waves (wave32)
#define GEMM_WAVES (NTILES * KSEG)   // 5504 wave-tasks
#define GEMM_GRID  (GEMM_WAVES / (GEMM_BLOCK / 32)) // 1376 blocks

// ---------------------------------------------------------------------------
// Pass 1: sum |W| (also pulls the whole 180MB W through L2, which fits 192MB)
// ---------------------------------------------------------------------------
__global__ void __launch_bounds__(256)
k_abs_sum(const float* __restrict__ w, double* __restrict__ sumAbs, int n4) {
    int i      = blockIdx.x * blockDim.x + threadIdx.x;
    int stride = gridDim.x * blockDim.x;
    const float4* w4 = (const float4*)w;
    float s = 0.0f;
    for (int j = i; j < n4; j += stride) {
        float4 v = w4[j];
        s += fabsf(v.x) + fabsf(v.y) + fabsf(v.z) + fabsf(v.w);
    }
    #pragma unroll
    for (int off = 16; off > 0; off >>= 1)
        s += __shfl_down(s, off, 32);
    if ((threadIdx.x & 31) == 0)
        atomicAdd(sumAbs, (double)s);   // global_atomic_add_f64
}

// ---------------------------------------------------------------------------
// Pass 1b: x fp32 -> f16 staging (512KB read, trivial)
// ---------------------------------------------------------------------------
__global__ void __launch_bounds__(256)
k_x_to_f16(const float* __restrict__ x, _Float16* __restrict__ xh, int n) {
    int i      = blockIdx.x * blockDim.x + threadIdx.x;
    int stride = gridDim.x * blockDim.x;
    for (; i < n; i += stride) xh[i] = (_Float16)x[i];
}

// ---------------------------------------------------------------------------
// Pass 2: fused ternarize + WMMA GEMM.  out += abs_mean * (x_f16 @ q^T)
// ---------------------------------------------------------------------------
__device__ __forceinline__ unsigned short tern_h(float v, float thr) {
    // exact f16 encoding of {-1, 0, +1}: 0xBC00 / 0x0000 / 0x3C00
    unsigned int sgn = __float_as_uint(v) & 0x80000000u;
    return (fabsf(v) > thr) ? (unsigned short)(0x3C00u | (sgn >> 16))
                            : (unsigned short)0;
}

// Build one ternarized 32x16 B tile slice (16 consecutive f16 per lane).
__device__ __forceinline__ v16h make_b_tile(const float* __restrict__ wp,
                                            float thr) {
    union { v16h h; v4u u[2]; } B;
    unsigned int pk[8];
    #pragma unroll
    for (int q = 0; q < 4; ++q) {
        float4 wv = *(const float4*)(wp + q * 4);
        unsigned int h0 = tern_h(wv.x, thr);
        unsigned int h1 = tern_h(wv.y, thr);
        unsigned int h2 = tern_h(wv.z, thr);
        unsigned int h3 = tern_h(wv.w, thr);
        pk[q * 2]     = h0 | (h1 << 16);
        pk[q * 2 + 1] = h2 | (h3 << 16);
    }
    B.u[0] = (v4u){pk[0], pk[1], pk[2], pk[3]};
    B.u[1] = (v4u){pk[4], pk[5], pk[6], pk[7]};
    return B.h;
}

__global__ void __launch_bounds__(GEMM_BLOCK)
k_gemm_wmma(const _Float16* __restrict__ xh, const float* __restrict__ w,
            const double* __restrict__ sumAbs, float* __restrict__ out) {
    const int lane = threadIdx.x & 31;

    // Wave-uniform task id forced into an SGPR: scalar loop bounds / bases,
    // no EXEC-masked loop, EXEC stays all-ones for every WMMA.
    const int waveId = __builtin_amdgcn_readfirstlane(
        blockIdx.x * (GEMM_BLOCK / 32) + (threadIdx.x >> 5));
    const int nt   = waveId % NTILES;          // column tile (16 outputs)
    const int kseg = waveId / NTILES;          // K segment
    const int o0   = nt * 16;
    const int kbeg = kseg * KSEG_LEN;

    const float mean  = (float)(*sumAbs) * (1.0f / 45088768.0f);
    const float scale = fmaxf(mean, 1e-5f);    // clamped abs-mean
    const float thr   = 0.7f * scale;

    const int ln = lane & 15;                  // m (A) / n (B,C)
    const int kh = lane >> 4;                  // k-half (A,B) / m-half (C)

    const _Float16* a0p = xh + (size_t)ln * IN_F + 16 * kh;  // token tile 0
    const _Float16* a1p = a0p + (size_t)16 * IN_F;           // token tile 1
    const float*    bp  = w + (size_t)(o0 + ln) * IN_F + 16 * kh; // W row

    v8f acc0 = {};
    v8f acc1 = {};

    // 64 K per trip -> 4 WMMAs per branch, two independent convert chains.
    for (int kc = kbeg; kc < kbeg + KSEG_LEN; kc += 64) {
        union { v16h h; v4u u[2]; } A0a, A1a, A0b, A1b;
        // A layout: elements 0-15 <- k0+16kh+0..15
        A0a.u[0] = *(const v4u*)(a0p + kc);
        A0a.u[1] = *(const v4u*)(a0p + kc + 8);
        A1a.u[0] = *(const v4u*)(a1p + kc);
        A1a.u[1] = *(const v4u*)(a1p + kc + 8);
        A0b.u[0] = *(const v4u*)(a0p + kc + 32);
        A0b.u[1] = *(const v4u*)(a0p + kc + 40);
        A1b.u[0] = *(const v4u*)(a1p + kc + 32);
        A1b.u[1] = *(const v4u*)(a1p + kc + 40);

        const float* wp = bp + kc;
        __builtin_prefetch(wp + 64, 0, 0);     // next trip's W chunk

        v16h Ba = make_b_tile(wp,      thr);
        v16h Bb = make_b_tile(wp + 32, thr);

        acc0 = __builtin_amdgcn_wmma_f32_16x16x32_f16(
                   false, A0a.h, false, Ba, (short)0, acc0, false, false);
        acc1 = __builtin_amdgcn_wmma_f32_16x16x32_f16(
                   false, A1a.h, false, Ba, (short)0, acc1, false, false);
        acc0 = __builtin_amdgcn_wmma_f32_16x16x32_f16(
                   false, A0b.h, false, Bb, (short)0, acc0, false, false);
        acc1 = __builtin_amdgcn_wmma_f32_16x16x32_f16(
                   false, A1b.h, false, Bb, (short)0, acc1, false, false);
    }

    // ---- C layout: lane=(n, mhalf), VGPR r -> m = r + 8*mhalf
    float* orow = out + o0 + ln;
    #pragma unroll
    for (int r = 0; r < 8; ++r) {
        const int m = r + 8 * kh;
        atomicAdd(orow + (size_t)m * OUT_F,        scale * acc0[r]);
        atomicAdd(orow + (size_t)(16 + m) * OUT_F, scale * acc1[r]);
    }
}

// ---------------------------------------------------------------------------
extern "C" void kernel_launch(void* const* d_in, const int* in_sizes, int n_in,
                              void* d_out, int out_size, void* d_ws, size_t ws_size,
                              hipStream_t stream) {
    const float* x = (const float*)d_in[0];   // [32, 4096]
    const float* w = (const float*)d_in[1];   // [11008, 4096]
    float* out = (float*)d_out;               // [32, 11008]

    double*   sumAbs = (double*)d_ws;                          // 8 bytes
    _Float16* xh     = (_Float16*)((char*)d_ws + 16);          // 256 KB

    hipMemsetAsync(d_ws, 0, 16, stream);
    hipMemsetAsync(d_out, 0, (size_t)out_size * sizeof(float), stream);

    k_abs_sum <<<2048, 256, 0, stream>>>(w, sumAbs, (IN_F * OUT_F) / 4);
    k_x_to_f16<<<128, 256, 0, stream>>>(x, xh, TOK * IN_F);
    k_gemm_wmma<<<GEMM_GRID, GEMM_BLOCK, 0, stream>>>(xh, w, sumAbs, out);
}
